// VanillaRNN_274877907384
// MI455X (gfx1250) — matrix operand order repaired and checked by reading, not compiled, not run on hardware
//
#include <hip/hip_runtime.h>
#include <hip/hip_bf16.h>
#include <math.h>

// ---- problem dims (fixed by reference) ----
#define T_STEPS 256
#define DIM_IN  1024
#define DIM_HID 1024
#define DIM_OUT 1024
#define DIM_B   128

typedef __attribute__((ext_vector_type(16))) __bf16 v16bf;
typedef __attribute__((ext_vector_type(8)))  __bf16 v8bf;
typedef __attribute__((ext_vector_type(8)))  float  v8f;

union afrag { v16bf v; v8bf h[2]; };

#define WMMA_BF16(a, b, c) \
    __builtin_amdgcn_wmma_f32_16x16x32_bf16(false, (a), false, (b), (short)0, (c), false, false)

// -----------------------------------------------------------------------------
// Wave-level bf16 GEMM accumulator: C(16x16, f32) += A(16xK) * Bt(16xK)^T
//   A  : row-major [M, lda] bf16  (A-fragment: lane=row, 2x contiguous 16B)
//   Bt : row-major [N, ldb] bf16  (transposed activations; lane=col, 32B row)
// K must be a multiple of 32; lda/ldb multiples of 16 (here 1024).
// -----------------------------------------------------------------------------
__device__ __forceinline__ v8f wmma_gemm_bf16(const __bf16* __restrict__ A, int lda,
                                              const __bf16* __restrict__ Bt, int ldb,
                                              int m0, int n0, int K, v8f acc)
{
    const int lane = threadIdx.x & 31;
    const int half = lane >> 4;                 // 0: lanes 0-15, 1: lanes 16-31
    const __bf16* arow = A  + (size_t)(m0 + (lane & 15)) * lda;
    const __bf16* brow = Bt + (size_t)(n0 + (lane & 15)) * ldb;

    for (int k0 = 0; k0 < K; k0 += 32) {
        const int kA = k0 + half * 8;           // A: elems 0-7 = K[kA..kA+7], 8-15 = K[kA+16..]
        const int kB = k0 + half * 8;           // B: elems 0-7 = K[kB..kB+7], 8-15 = K[kB+16..]
        afrag a, b;
        a.h[0] = *(const v8bf*)(arow + kA);
        a.h[1] = *(const v8bf*)(arow + kA + 16);
        b.h[0] = *(const v8bf*)(brow + kB);
        b.h[1] = *(const v8bf*)(brow + kB + 16);
        if (k0 + 128 < K) {                     // global_prefetch_b8 one block ahead
            __builtin_prefetch(arow + kA + 128, 0, 1);
            __builtin_prefetch(brow + kB + 128, 0, 1);
        }
        acc = WMMA_BF16(a.v, b.v, acc);
    }
    return acc;
}

// ---- f32 -> bf16 elementwise (weights) ----
__global__ void cvt_f32_to_bf16(const float* __restrict__ in, __bf16* __restrict__ out, int n)
{
    int i = blockIdx.x * blockDim.x + threadIdx.x;
    if (i < n) out[i] = (__bf16)in[i];
}

// ---- f32 [S, R, C] -> bf16 [S, C, R] (transpose-convert activations) ----
__global__ void transpose_cvt_bf16(const float* __restrict__ in, __bf16* __restrict__ out,
                                   int R, int C, long long total)
{
    long long i = (long long)blockIdx.x * blockDim.x + threadIdx.x;
    if (i >= total) return;
    int r = (int)(i % R);
    long long j = i / R;
    int c = (int)(j % C);
    long long s = j / C;
    out[(s * C + c) * (long long)R + r] = (__bf16)in[(s * R + r) * (long long)C + c];
}

// -----------------------------------------------------------------------------
// One recurrence step: h_{t+1} = tanh(Wxh@x_t + Whh@h_t + bh)
// 512 tiles (64 x 8) over [HID, B]; 4 waves/block -> 128 blocks.
// One 16x16 tile per wave: the 256-step chain is latency-bound, so maximize
// SIMD spread rather than per-wave intensity (weights are L2-resident).
// Writes h transposed-bf16 into Ht[t+1]; last step also writes f32 h_final.
// -----------------------------------------------------------------------------
__global__ __launch_bounds__(128)
void rnn_step_kernel(const __bf16* __restrict__ Wxh,   // [HID, IN] bf16
                     const __bf16* __restrict__ Whh,   // [HID, HID] bf16
                     const __bf16* __restrict__ Xt,    // [T, B, IN] bf16
                     __bf16* __restrict__ Ht,          // [T+1, B, HID] bf16
                     const float* __restrict__ bh,     // [HID]
                     float* __restrict__ h_final,      // [HID, B] (d_out tail)
                     int t, int is_last)
{
    const int wave = threadIdx.x >> 5;
    const int tile = blockIdx.x * (blockDim.x >> 5) + wave;   // 0..511
    const int m0 = (tile >> 3) * 16;                          // HID tile
    const int n0 = (tile & 7) * 16;                           // B tile

    v8f acc = {};
    acc = wmma_gemm_bf16(Wxh, DIM_IN,  Xt + (size_t)t * DIM_B * DIM_IN,  DIM_IN,
                         m0, n0, DIM_IN,  acc);
    acc = wmma_gemm_bf16(Whh, DIM_HID, Ht + (size_t)t * DIM_B * DIM_HID, DIM_HID,
                         m0, n0, DIM_HID, acc);

    const int lane = threadIdx.x & 31;
    const int half = lane >> 4;
    const int n = n0 + (lane & 15);
    __bf16* hout = Ht + (size_t)(t + 1) * DIM_B * DIM_HID + (size_t)n * DIM_HID;

#pragma unroll
    for (int v = 0; v < 8; ++v) {
        const int m = m0 + v + half * 8;
        const float h = tanhf(acc[v] + bh[m]);
        hout[m] = (__bf16)h;                       // transposed bf16 for next step's B operand
        if (is_last) h_final[(size_t)m * DIM_B + n] = h;
    }
}

// -----------------------------------------------------------------------------
// Output projection for all t: ys[t] = Why @ h_{t+1} + by
// Each wave computes a 64x16 strip: 4 M-subtiles share one B fragment per
// K-iter (10 b128 loads feed 4 WMMAs instead of 16), raising L2 intensity.
// strips = T * 16 * 8 = 32768 ; 4 waves/block -> 8192 blocks.
// -----------------------------------------------------------------------------
__global__ __launch_bounds__(128)
void rnn_out_kernel(const __bf16* __restrict__ Why,    // [OUT, HID] bf16
                    const __bf16* __restrict__ Ht,     // [T+1, B, HID] bf16
                    const float* __restrict__ by,      // [OUT]
                    float* __restrict__ ys)            // [T, OUT, B]
{
    const int wave = threadIdx.x >> 5;
    const int tile = blockIdx.x * (blockDim.x >> 5) + wave;   // 0 .. 32767
    const int n0 = (tile & 7) * 16;                           // B tile
    const int r  = tile >> 3;
    const int ms = (r & 15) * 64;                             // OUT strip base (64 rows)
    const int t  = r >> 4;

    const int lane = threadIdx.x & 31;
    const int half = lane >> 4;
    const __bf16* brow = Ht + (size_t)(t + 1) * DIM_B * DIM_HID
                            + (size_t)(n0 + (lane & 15)) * DIM_HID;
    const __bf16* arow = Why + (size_t)(ms + (lane & 15)) * DIM_HID;

    v8f acc0 = {}, acc1 = {}, acc2 = {}, acc3 = {};
    for (int k0 = 0; k0 < DIM_HID; k0 += 32) {
        const int kA = k0 + half * 8;
        const int kB = k0 + half * 8;
        afrag b, a0, a1, a2, a3;
        b.h[0]  = *(const v8bf*)(brow + kB);
        b.h[1]  = *(const v8bf*)(brow + kB + 16);
        a0.h[0] = *(const v8bf*)(arow + kA);
        a0.h[1] = *(const v8bf*)(arow + kA + 16);
        a1.h[0] = *(const v8bf*)(arow + 16 * DIM_HID + kA);
        a1.h[1] = *(const v8bf*)(arow + 16 * DIM_HID + kA + 16);
        a2.h[0] = *(const v8bf*)(arow + 32 * DIM_HID + kA);
        a2.h[1] = *(const v8bf*)(arow + 32 * DIM_HID + kA + 16);
        a3.h[0] = *(const v8bf*)(arow + 48 * DIM_HID + kA);
        a3.h[1] = *(const v8bf*)(arow + 48 * DIM_HID + kA + 16);
        acc0 = WMMA_BF16(a0.v, b.v, acc0);
        acc1 = WMMA_BF16(a1.v, b.v, acc1);
        acc2 = WMMA_BF16(a2.v, b.v, acc2);
        acc3 = WMMA_BF16(a3.v, b.v, acc3);
    }

    const int n = n0 + (lane & 15);
    float* yrow = ys + (size_t)t * DIM_OUT * DIM_B;

#pragma unroll
    for (int v = 0; v < 8; ++v) {
        const int mv = v + half * 8;
        yrow[(size_t)(ms +  0 + mv) * DIM_B + n] = acc0[v] + by[ms +  0 + mv];
        yrow[(size_t)(ms + 16 + mv) * DIM_B + n] = acc1[v] + by[ms + 16 + mv];
        yrow[(size_t)(ms + 32 + mv) * DIM_B + n] = acc2[v] + by[ms + 32 + mv];
        yrow[(size_t)(ms + 48 + mv) * DIM_B + n] = acc3[v] + by[ms + 48 + mv];
    }
}

// -----------------------------------------------------------------------------
// Host launcher
// Inputs (setup_inputs order): inputs[T,IN,B], h_prev[HID,B], Wxh[HID,IN],
//                              Whh[HID,HID], Why[OUT,HID], bh[HID,1], by[OUT,1]
// d_out: ys[T,OUT,B] f32 followed by h_final[HID,B] f32.
// -----------------------------------------------------------------------------
extern "C" void kernel_launch(void* const* d_in, const int* in_sizes, int n_in,
                              void* d_out, int out_size, void* d_ws, size_t ws_size,
                              hipStream_t stream)
{
    (void)in_sizes; (void)n_in; (void)out_size; (void)ws_size;

    const float* x    = (const float*)d_in[0];
    const float* h0   = (const float*)d_in[1];
    const float* Wxh  = (const float*)d_in[2];
    const float* Whh  = (const float*)d_in[3];
    const float* Why  = (const float*)d_in[4];
    const float* bh   = (const float*)d_in[5];
    const float* by   = (const float*)d_in[6];

    float* ys      = (float*)d_out;
    float* h_final = ys + (size_t)T_STEPS * DIM_OUT * DIM_B;

    // Workspace carve-up (bf16): weights 3 x 2MB, Xt 64MB, Ht 64.25MB (~134.5MB total)
    char* ws = (char*)d_ws;
    __bf16* Wxhb = (__bf16*)ws;  ws += (size_t)DIM_HID * DIM_IN  * sizeof(__bf16);
    __bf16* Whhb = (__bf16*)ws;  ws += (size_t)DIM_HID * DIM_HID * sizeof(__bf16);
    __bf16* Whyb = (__bf16*)ws;  ws += (size_t)DIM_OUT * DIM_HID * sizeof(__bf16);
    __bf16* Xt   = (__bf16*)ws;  ws += (size_t)T_STEPS * DIM_B * DIM_IN * sizeof(__bf16);
    __bf16* Ht   = (__bf16*)ws;  // [T+1, B, HID]

    // 1) weight conversions (parallel, independent)
    {
        const int n = DIM_HID * DIM_IN;
        cvt_f32_to_bf16<<<n / 256, 256, 0, stream>>>(Wxh, Wxhb, n);
        cvt_f32_to_bf16<<<n / 256, 256, 0, stream>>>(Whh, Whhb, DIM_HID * DIM_HID);
        cvt_f32_to_bf16<<<n / 256, 256, 0, stream>>>(Why, Whyb, DIM_OUT * DIM_HID);
    }

    // 2) transpose-convert activations: X -> Xt [T,B,IN], h_prev -> Ht slot 0
    {
        const long long totX = (long long)T_STEPS * DIM_IN * DIM_B;
        transpose_cvt_bf16<<<(int)(totX / 256), 256, 0, stream>>>(x, Xt, DIM_IN, DIM_B, totX);
        const long long totH = (long long)DIM_HID * DIM_B;
        transpose_cvt_bf16<<<(int)(totH / 256), 256, 0, stream>>>(h0, Ht, DIM_HID, DIM_B, totH);
    }

    // 3) serial recurrence: 256 dependent step GEMM kernels (graph replay amortizes launch)
    for (int t = 0; t < T_STEPS; ++t) {
        rnn_step_kernel<<<128, 128, 0, stream>>>(Wxhb, Whhb, Xt, Ht, bh, h_final,
                                                 t, t == T_STEPS - 1);
    }

    // 4) fully parallel output projection for all timesteps (64x16 per wave)
    rnn_out_kernel<<<8192, 128, 0, stream>>>(Whyb, Ht, by, ys);
}
